// Predictor_10771777978780
// MI455X (gfx1250) — matrix operand order repaired and checked by reading, not compiled, not run on hardware
//
#include <hip/hip_runtime.h>
#include <hip/hip_bf16.h>
#include <math.h>

typedef __attribute__((ext_vector_type(16))) __bf16 v16bf;
typedef __attribute__((ext_vector_type(8)))  __bf16 v8bf;
typedef __attribute__((ext_vector_type(8)))  float  v8f;

#define VOCAB   27
#define NEMBD   512
#define DMODEL  256
#define GATES   1024
#define BATCH   256
#define TSTEPS  512
#define PAD_IDX 26

// ---- workspace layout (bytes, all 256B aligned) ----
#define WS_XG0   0                                  // f32 [27][1024] : W_ih0@emb + b_ih0 + b_hh0
#define WS_WHH0  (WS_XG0 + VOCAB*GATES*4)           // bf16 [1024][256]
#define WS_W1CAT (WS_WHH0 + GATES*DMODEL*2)         // bf16 [1024][512] : [W_ih1 | W_hh1]
#define WS_B1SUM (WS_W1CAT + GATES*2*DMODEL*2)      // f32 [1024] : b_ih1 + b_hh1
#define WS_H1    (WS_B1SUM + GATES*4)               // bf16 [16 tiles][512 t][16 b][256 h]

__device__ __forceinline__ float sigm(float x) { return 1.0f / (1.0f + __expf(-x)); }

// A fragment (16x32 bf16) from row-major LDS buffer, per ISA VGPR layout:
// lanes 0-15: K = k0+0..7 and k0+16..23 ; lanes 16-31: K = k0+8..15 and k0+24..31
__device__ __forceinline__ v16bf load_a_frag(const __bf16* base, int ldk, int lane, int k0) {
  int m  = lane & 15;
  int kb = k0 + ((lane >> 4) << 3);
  const __bf16* p = base + m * ldk + kb;
  v8bf lo = *(const v8bf*)(p);
  v8bf hi = *(const v8bf*)(p + 16);
  return __builtin_shufflevector(lo, hi, 0,1,2,3,4,5,6,7,8,9,10,11,12,13,14,15);
}

// B fragment (32x16 bf16) straight from row-major W[gate][k]:
// lanes 0-15: K = k0+0..7 and k0+16..23 ; lanes 16-31: K = k0+8..15 and k0+24..31 (matches A).
__device__ __forceinline__ v16bf load_b_frag(const __bf16* W, int ldk, int lane, int ncol, int k0) {
  int n  = ncol + (lane & 15);
  int kk = k0 + ((lane >> 4) << 3);
  const __bf16* p = W + n * ldk + kk;
  v8bf lo = *(const v8bf*)(p);
  v8bf hi = *(const v8bf*)(p + 16);
  return __builtin_shufflevector(lo, hi, 0,1,2,3,4,5,6,7,8,9,10,11,12,13,14,15);
}

// -------- prep: fold embedding into gate table, bf16 weight copies, bias sums --------
__global__ void prep_kernel(const float* __restrict__ emb,  const float* __restrict__ w_ih0,
                            const float* __restrict__ b_ih0, const float* __restrict__ b_hh0,
                            const float* __restrict__ w_ih1, const float* __restrict__ w_hh1,
                            const float* __restrict__ b_ih1, const float* __restrict__ b_hh1,
                            const float* __restrict__ w_hh0, char* __restrict__ ws) {
  int idx = blockIdx.x * blockDim.x + threadIdx.x;
  float*  xg0  = (float*) (ws + WS_XG0);
  __bf16* whh0 = (__bf16*)(ws + WS_WHH0);
  __bf16* w1c  = (__bf16*)(ws + WS_W1CAT);
  float*  b1s  = (float*) (ws + WS_B1SUM);

  if (idx < VOCAB * GATES) {
    int v = idx / GATES, g = idx % GATES;
    float acc = b_ih0[g] + b_hh0[g];
    if (v != PAD_IDX) {
      const float* e = emb + v * NEMBD;
      const float* w = w_ih0 + g * NEMBD;
      #pragma unroll 8
      for (int k = 0; k < NEMBD; ++k) acc = fmaf(e[k], w[k], acc);
    }
    xg0[v * GATES + g] = acc;
  } else if (idx < VOCAB * GATES + GATES * DMODEL) {
    int i = idx - VOCAB * GATES;
    whh0[i] = (__bf16)w_hh0[i];
  } else if (idx < VOCAB * GATES + GATES * DMODEL + GATES * 2 * DMODEL) {
    int i = idx - VOCAB * GATES - GATES * DMODEL;
    int g = i / (2 * DMODEL), k = i % (2 * DMODEL);
    float v = (k < DMODEL) ? w_ih1[g * DMODEL + k] : w_hh1[g * DMODEL + (k - DMODEL)];
    w1c[i] = (__bf16)v;
  } else if (idx < VOCAB * GATES + GATES * DMODEL + GATES * 2 * DMODEL + GATES) {
    int g = idx - (VOCAB * GATES + GATES * DMODEL + GATES * 2 * DMODEL);
    b1s[g] = b_ih1[g] + b_hh1[g];
  }
}

// -------- persistent fused 2-layer LSTM + head; one WG = one 16-row batch tile --------
__global__ __launch_bounds__(512, 1)
void lstm_persistent_kernel(const int* __restrict__ ids,
                            const float* __restrict__ fc1_w, const float* __restrict__ fc1_b,
                            const float* __restrict__ fc2_w, const float* __restrict__ fc2_b,
                            char* __restrict__ ws, float* __restrict__ out) {
  __shared__ __align__(16) __bf16 lds_h[16 * 512];   // [16 rows][ 0:256 = x-part | 256:512 = h-part ]

  const int tid  = threadIdx.x;
  const int lane = tid & 31;
  const int wv   = tid >> 5;            // wave 0..15, owns hidden slice [16*wv, 16*wv+16)
  const int blk  = blockIdx.x;          // batch tile 0..15
  const int b0   = blk << 4;

  const float*  xg0  = (const float*) (ws + WS_XG0);
  const __bf16* whh0 = (const __bf16*)(ws + WS_WHH0);
  const __bf16* w1c  = (const __bf16*)(ws + WS_W1CAT);
  const float*  b1s  = (const float*) (ws + WS_B1SUM);
  __bf16*       h1   = (__bf16*)      (ws + WS_H1);

  const int col = (wv << 4) + (lane & 15);   // hidden/gate column within a 256-wide gate block
  const int rof = (lane >> 4) << 3;          // D-fragment row offset (0 or 8)

  // zero h state buffer (16 KB)
  {
    v8bf z = {};
    ((v8bf*)lds_h)[tid * 2]     = z;
    ((v8bf*)lds_h)[tid * 2 + 1] = z;
  }
  __syncthreads();

  float c[8];
  #pragma unroll
  for (int v = 0; v < 8; ++v) c[v] = 0.0f;

  // =================== Layer 0: g = xg0[token] + h @ W_hh0^T ===================
  for (int t = 0; t < TSTEPS; ++t) {
    v8f ai = {}, af = {}, ag = {}, ao = {};
    #pragma unroll
    for (int k0 = 0; k0 < DMODEL; k0 += 32) {
      v16bf a  = load_a_frag(lds_h, 512, lane, k0);
      v16bf bI = load_b_frag(whh0, DMODEL, lane, 0   + (wv << 4), k0);
      v16bf bF = load_b_frag(whh0, DMODEL, lane, 256 + (wv << 4), k0);
      v16bf bG = load_b_frag(whh0, DMODEL, lane, 512 + (wv << 4), k0);
      v16bf bO = load_b_frag(whh0, DMODEL, lane, 768 + (wv << 4), k0);
      ai = __builtin_amdgcn_wmma_f32_16x16x32_bf16(false, a, false, bI, (short)0, ai, false, false);
      af = __builtin_amdgcn_wmma_f32_16x16x32_bf16(false, a, false, bF, (short)0, af, false, false);
      ag = __builtin_amdgcn_wmma_f32_16x16x32_bf16(false, a, false, bG, (short)0, ag, false, false);
      ao = __builtin_amdgcn_wmma_f32_16x16x32_bf16(false, a, false, bO, (short)0, ao, false, false);
    }
    __syncthreads();   // all A reads done before overwriting h

    __bf16* gout = h1 + (((size_t)(blk * TSTEPS + t)) << 12);  // 16*256 bf16 per step
    #pragma unroll
    for (int v = 0; v < 8; ++v) {
      int r   = v + rof;
      int tok = ids[(b0 + r) * TSTEPS + t];
      const float* xg = xg0 + tok * GATES + col;
      float gi = ai[v] + xg[0];
      float gf = af[v] + xg[256];
      float gg = ag[v] + xg[512];
      float go = ao[v] + xg[768];
      float iv = sigm(gi), fv = sigm(gf), gv = tanhf(gg), ov = sigm(go);
      c[v] = fv * c[v] + iv * gv;
      float hv = ov * tanhf(c[v]);
      __bf16 hb = (__bf16)hv;
      lds_h[r * 512 + col] = hb;           // h state (cols 0..255 this phase)
      gout[(r << 8) + col] = hb;           // stream h1 for layer 1
    }
    __syncthreads();
  }

  // reset state for layer 1; h lives in cols [256,512) now
  #pragma unroll
  for (int v = 0; v < 8; ++v) c[v] = 0.0f;
  {
    v8bf z = {};
    int r = tid >> 5, coff = 256 + ((tid & 31) << 3);
    *(v8bf*)(lds_h + r * 512 + coff) = z;
  }

  // ============ Layer 1: g = [h1_t | h] @ [W_ih1 | W_hh1]^T + b1sum (K=512) ============
  const __bf16* h1src = h1 + (((size_t)blk * TSTEPS) << 12);
  for (int t = 0; t < TSTEPS; ++t) {
    // cooperative load of h1_t into cols [0,256): 2x v8bf per thread
    {
      const v8bf* src = (const v8bf*)(h1src + ((size_t)t << 12));
      #pragma unroll
      for (int j = 0; j < 2; ++j) {
        int q = tid * 2 + j;
        int r = q >> 5, coff = (q & 31) << 3;
        *(v8bf*)(lds_h + r * 512 + coff) = src[q];
      }
      __builtin_prefetch((const void*)(h1src + (((size_t)(t + 1)) << 12) + tid * 8), 0, 0);
    }
    __syncthreads();   // h1_t and h_{t-1} visible

    v8f ai = {}, af = {}, ag = {}, ao = {};
    #pragma unroll
    for (int k0 = 0; k0 < 2 * DMODEL; k0 += 32) {
      v16bf a  = load_a_frag(lds_h, 512, lane, k0);
      v16bf bI = load_b_frag(w1c, 2 * DMODEL, lane, 0   + (wv << 4), k0);
      v16bf bF = load_b_frag(w1c, 2 * DMODEL, lane, 256 + (wv << 4), k0);
      v16bf bG = load_b_frag(w1c, 2 * DMODEL, lane, 512 + (wv << 4), k0);
      v16bf bO = load_b_frag(w1c, 2 * DMODEL, lane, 768 + (wv << 4), k0);
      ai = __builtin_amdgcn_wmma_f32_16x16x32_bf16(false, a, false, bI, (short)0, ai, false, false);
      af = __builtin_amdgcn_wmma_f32_16x16x32_bf16(false, a, false, bF, (short)0, af, false, false);
      ag = __builtin_amdgcn_wmma_f32_16x16x32_bf16(false, a, false, bG, (short)0, ag, false, false);
      ao = __builtin_amdgcn_wmma_f32_16x16x32_bf16(false, a, false, bO, (short)0, ao, false, false);
    }
    __syncthreads();   // all reads done before state update

    float bi = b1s[col], bf = b1s[256 + col], bg = b1s[512 + col], bo = b1s[768 + col];
    #pragma unroll
    for (int v = 0; v < 8; ++v) {
      int r = v + rof;
      float iv = sigm(ai[v] + bi), fv = sigm(af[v] + bf);
      float gv = tanhf(ag[v] + bg), ov = sigm(ao[v] + bo);
      c[v] = fv * c[v] + iv * gv;
      float hv = ov * tanhf(c[v]);
      lds_h[r * 512 + 256 + col] = (__bf16)hv;
    }
  }
  __syncthreads();

  // =================== Head: logits = relu(h2_last @ fc1^T + b1) @ fc2^T + b2 ===================
  if (tid < 16) {
    int m = tid;
    float logit = fc2_b[0];
    for (int u = 0; u < 128; ++u) {
      float s = fc1_b[u];
      const float* wr = fc1_w + u * 256;
      #pragma unroll 8
      for (int j = 0; j < 256; ++j)
        s = fmaf((float)lds_h[m * 512 + 256 + j], wr[j], s);
      s = fmaxf(s, 0.0f);
      logit = fmaf(s, fc2_w[u], logit);
    }
    out[b0 + m] = logit;
  }
}

extern "C" void kernel_launch(void* const* d_in, const int* in_sizes, int n_in,
                              void* d_out, int out_size, void* d_ws, size_t ws_size,
                              hipStream_t stream) {
  const int*   ids   = (const int*)  d_in[0];
  const float* emb   = (const float*)d_in[1];
  const float* w_ih0 = (const float*)d_in[2];
  const float* w_hh0 = (const float*)d_in[3];
  const float* b_ih0 = (const float*)d_in[4];
  const float* b_hh0 = (const float*)d_in[5];
  const float* w_ih1 = (const float*)d_in[6];
  const float* w_hh1 = (const float*)d_in[7];
  const float* b_ih1 = (const float*)d_in[8];
  const float* b_hh1 = (const float*)d_in[9];
  const float* fc1_w = (const float*)d_in[10];
  const float* fc1_b = (const float*)d_in[11];
  const float* fc2_w = (const float*)d_in[12];
  const float* fc2_b = (const float*)d_in[13];
  char*  ws  = (char*)d_ws;
  float* out = (float*)d_out;

  // prep work items: 27*1024 + 1024*256 + 1024*512 + 1024 = 815104 = 3184 * 256
  prep_kernel<<<3184, 256, 0, stream>>>(emb, w_ih0, b_ih0, b_hh0,
                                        w_ih1, w_hh1, b_ih1, b_hh1, w_hh0, ws);
  lstm_persistent_kernel<<<16, 512, 0, stream>>>(ids, fc1_w, fc1_b, fc2_w, fc2_b, ws, out);
}
